// MultiheadChannelAttention_4896262717577
// MI455X (gfx1250) — matrix operand order repaired and checked by reading, not compiled, not run on hardware
//
#include <hip/hip_runtime.h>
#include <hip/hip_bf16.h>
#include <cstdint>
#include <cstddef>

// ---------------- problem constants ----------------
constexpr int Bn  = 8;
constexpr int Cn  = 128;
constexpr int Hn  = 128;
constexpr int Wn  = 128;
constexpr int HWn = Hn * Wn;        // 16384
constexpr int HEADS = 8;
constexpr int Dn  = HWn / HEADS;    // 2048 (per-head spatial dim)
constexpr int KW  = Cn * 9;         // 1152 (implicit-GEMM K for 3x3 conv)
constexpr int NKC = Dn / 32;        // 64 k-chunks in attention phase 1

// ---------------- CDNA5 WMMA types ----------------
typedef __attribute__((ext_vector_type(16))) __bf16 v16bf;
typedef __attribute__((ext_vector_type(8)))  float  v8f;
static_assert(sizeof(v16bf) == 32, "v16bf must be 32B");
static_assert(sizeof(v8f) == 32, "v8f must be 32B");

__device__ inline v8f zero8() {
  v8f z;
#pragma unroll
  for (int i = 0; i < 8; ++i) z[i] = 0.0f;
  return z;
}

// fp32 -> bf16 round-to-nearest-even
__device__ inline unsigned short f2bf(float f) {
  union { float f; unsigned int u; } x; x.f = f;
  unsigned int u = x.u;
  unsigned int r = u + 0x7fffu + ((u >> 16) & 1u);
  return (unsigned short)(r >> 16);
}

// A-fragment (16x32 bf16): per lane two 16B chunks, 16 elements apart
// (K off {0..7,16..23} for lanes 0-15; {8..15,24..31} for lanes 16-31 via +hi*8)
__device__ inline v16bf ld_frag_a(const unsigned short* p) {
  union { uint4 q[2]; v16bf v; } u;
  u.q[0] = *(const uint4*)(p);
  u.q[1] = *(const uint4*)(p + 16);
  return u.v;
}
// B-fragment (32x16 bf16): per lane two 16B chunks, 16 elements apart (same K order as A-frag)
__device__ inline v16bf ld_frag_b(const unsigned short* p) {
  union { uint4 q[2]; v16bf v; } u;
  u.q[0] = *(const uint4*)(p);
  u.q[1] = *(const uint4*)(p + 16);
  return u.v;
}

__device__ inline v8f wmma_bf16(v16bf a, v16bf b, v8f c) {
  return __builtin_amdgcn_wmma_f32_16x16x32_bf16(
      /*neg_a=*/false, a, /*neg_b=*/false, b,
      /*c_mod=*/(short)0, c, /*reuse_a=*/false, /*reuse_b=*/false);
}

// ---------------- CDNA5 async global->LDS copy (ASYNCcnt path) ----------------
// GV addressing: lds[VGPR_dst + offset] <- mem[VGPR_addr64 + offset], 16B/lane.
__device__ inline void async_copy_b128(uint32_t lds_byte_off, const void* gaddr) {
  asm volatile("global_load_async_to_lds_b128 %0, %1, off"
               :: "v"(lds_byte_off), "v"((unsigned long long)(uintptr_t)gaddr)
               : "memory");
}
__device__ inline void wait_async_le4() {
  asm volatile("s_wait_asynccnt 0x4" ::: "memory");
}
__device__ inline void wait_async_0() {
  asm volatile("s_wait_asynccnt 0x0" ::: "memory");
}

// =====================================================================
// Kernel 0: convert 3x3 conv weights fp32 -> bf16 (layout unchanged:
// [co][ci][ky][kx] => K = ci*9 + t is contiguous -> direct A-frag loads)
// =====================================================================
__global__ void __launch_bounds__(256) cvt_w_kernel(const float* __restrict__ w,
                                                    unsigned short* __restrict__ o,
                                                    int n) {
  int i = blockIdx.x * 256 + threadIdx.x;
  if (i < n) o[i] = f2bf(w[i]);
}

// =====================================================================
// Kernel 1: fused depthwise 9x9 convs -> q,k,v (bf16). One block per
// (b, c, 32x32 tile); x tile + halo staged in LDS, read x once.
// =====================================================================
__global__ void __launch_bounds__(256) dwconv_qkv_kernel(
    const float* __restrict__ x,
    const float* __restrict__ wq, const float* __restrict__ bq,
    const float* __restrict__ wk, const float* __restrict__ bk,
    const float* __restrict__ wv, const float* __restrict__ bv,
    unsigned short* __restrict__ qo, unsigned short* __restrict__ ko,
    unsigned short* __restrict__ vo) {
  const int tid = threadIdx.x;
  const int tile = blockIdx.x;            // 0..15
  const int c = blockIdx.y;
  const int b = blockIdx.z;
  const int ty0 = (tile >> 2) * 32;
  const int tx0 = (tile & 3) * 32;

  __shared__ float xs[40][41];
  __shared__ float wqs[81], wks[81], wvs[81];
  if (tid < 81) {
    wqs[tid] = wq[c * 81 + tid];
    wks[tid] = wk[c * 81 + tid];
    wvs[tid] = wv[c * 81 + tid];
  }
  const float* xp = x + (size_t)(b * Cn + c) * HWn;
  for (int i = tid; i < 40 * 40; i += 256) {
    int ly = i / 40, lx = i - ly * 40;
    int gy = ty0 + ly - 4, gx = tx0 + lx - 4;
    float v0 = 0.0f;
    if (gy >= 0 && gy < Hn && gx >= 0 && gx < Wn) v0 = xp[gy * Wn + gx];
    xs[ly][lx] = v0;
  }
  __syncthreads();

  const float bqv = bq[c], bkv = bk[c], bvv = bv[c];
  unsigned short* qp = qo + (size_t)(b * Cn + c) * HWn;
  unsigned short* kp = ko + (size_t)(b * Cn + c) * HWn;
  unsigned short* vp = vo + (size_t)(b * Cn + c) * HWn;

  for (int p = 0; p < 4; ++p) {
    int idx = tid + p * 256;
    int oy = idx >> 5, ox = idx & 31;
    float aq = 0.f, ak = 0.f, av = 0.f;
#pragma unroll
    for (int ky = 0; ky < 9; ++ky) {
#pragma unroll
      for (int kx = 0; kx < 9; ++kx) {
        float xv = xs[oy + ky][ox + kx];
        int t = ky * 9 + kx;
        aq = fmaf(xv, wqs[t], aq);
        ak = fmaf(xv, wks[t], ak);
        av = fmaf(xv, wvs[t], av);
      }
    }
    int hw = (ty0 + oy) * Wn + tx0 + ox;
    qp[hw] = f2bf(aq + bqv);
    kp[hw] = f2bf(ak + bkv);
    vp[hw] = f2bf(av + bvv);
  }
}

// =====================================================================
// Kernel 2: channel attention per (b, head). 8 waves; wave w owns rows
// 16w..16w+15. Phase 1: S = Q*K^T (bf16 WMMA, f32 acc), Q/K chunks
// double-buffered in LDS via global_load_async_to_lds_b128 (ASYNCcnt).
// Phase-1 loop kept unroll(1)+peeled so the eight v8f accumulators get
// stable register homes (no v_mov/v_nop hazard churn around the WMMAs).
// Softmax is wave-local (half-wave shfl_xor). Phase 2: ctx = P*V with
// P (bf16) in LDS and V staged transposed (vectorized b128 loads,
// packed b32 LDS scatter).
// =====================================================================
__global__ void __launch_bounds__(256) attn_kernel(
    const unsigned short* __restrict__ q, const unsigned short* __restrict__ k,
    const unsigned short* __restrict__ v, unsigned short* __restrict__ ctx) {
  const int tid = threadIdx.x;
  const int wave = tid >> 5;
  const int lane = tid & 31;
  const int lhalf = lane & 15;
  const int hi = lane >> 4;   // 0: lanes 0-15, 1: lanes 16-31
  const int bh = blockIdx.x;
  const int b = bh >> 3, h = bh & 7;
  const size_t base = (size_t)b * Cn * HWn + (size_t)h * Dn;

  __shared__ __align__(32) unsigned char smem[65536];
  // phase 1: double-buffered Q/K chunks. buffer i at smem + i*16384:
  //   Q tile [128][32] bf16 (8KB) then K tile [128][32] bf16 (8KB)
  // phase 2: Ps [128][128] bf16 at smem (32KB), Vt [128][128] bf16 at +32768
  unsigned short* Ps = (unsigned short*)smem;
  unsigned short* Vt = (unsigned short*)(smem + 32768);

  // async staging: 512 x 16B segments per 8KB tile; thread covers segs
  // tid and tid+256 => per-instruction lane addresses are consecutive 16B.
  const int r0 = tid >> 2, p0 = tid & 3;      // seg tid
  const int r1 = r0 + 64;                     // seg tid+256 (same part)
  const unsigned short* qg = q + base;
  const unsigned short* kg = k + base;

  auto issue_chunk = [&](int kk, int buf) {
    const int k0 = kk * 32;
    uint32_t qd = (uint32_t)(uintptr_t)(smem + buf * 16384);
    uint32_t kd = qd + 8192;
    async_copy_b128(qd + r0 * 64 + p0 * 16, qg + (size_t)r0 * HWn + k0 + p0 * 8);
    async_copy_b128(qd + r1 * 64 + p0 * 16, qg + (size_t)r1 * HWn + k0 + p0 * 8);
    async_copy_b128(kd + r0 * 64 + p0 * 16, kg + (size_t)r0 * HWn + k0 + p0 * 8);
    async_copy_b128(kd + r1 * 64 + p0 * 16, kg + (size_t)r1 * HWn + k0 + p0 * 8);
  };

  v8f acc[8];
#pragma unroll
  for (int n = 0; n < 8; ++n) acc[n] = zero8();

  auto qk_step = [&](int buf) {
    const unsigned short* Qb = (const unsigned short*)(smem + buf * 16384);
    const unsigned short* Kb = Qb + 4096;   // +8192 bytes
    v16bf af = ld_frag_a(Qb + (wave * 16 + lhalf) * 32 + hi * 8);
#pragma unroll
    for (int nt = 0; nt < 8; ++nt) {
      v16bf bfr = ld_frag_b(Kb + (nt * 16 + lhalf) * 32 + hi * 8);
      acc[nt] = wmma_bf16(af, bfr, acc[nt]);
    }
  };

  // ---- Phase 1: S = Q * K^T, K-dim 2048, 32-wide double-buffered chunks ----
  issue_chunk(0, 0);
#pragma unroll 1
  for (int kk = 0; kk < NKC - 1; ++kk) {
    issue_chunk(kk + 1, (kk + 1) & 1);
    wait_async_le4();   // chunk kk's 4 ops (per wave, in-order) are done
    __syncthreads();    // all waves' async data visible
    qk_step(kk & 1);
    __syncthreads();    // fragment reads done before buffer is re-issued
  }
  wait_async_0();
  __syncthreads();
  qk_step((NKC - 1) & 1);
  __syncthreads();      // last reads done before Ps aliases the buffers

  // ---- Softmax (rows live in one VGPR slot across a 16-lane half) ----
  const float scale = 0.0078125f;  // 1/sqrt(H*W) = 1/128
  float rinv[8];
#pragma unroll
  for (int r = 0; r < 8; ++r) {
    float m = acc[0][r];
#pragma unroll
    for (int t = 1; t < 8; ++t) m = fmaxf(m, acc[t][r]);
    m = fmaxf(m, __shfl_xor(m, 1, 32));
    m = fmaxf(m, __shfl_xor(m, 2, 32));
    m = fmaxf(m, __shfl_xor(m, 4, 32));
    m = fmaxf(m, __shfl_xor(m, 8, 32));
    float s = 0.0f;
#pragma unroll
    for (int t = 0; t < 8; ++t) {
      float e = __expf((acc[t][r] - m) * scale);
      acc[t][r] = e;
      s += e;
    }
    s += __shfl_xor(s, 1, 32);
    s += __shfl_xor(s, 2, 32);
    s += __shfl_xor(s, 4, 32);
    s += __shfl_xor(s, 8, 32);
    rinv[r] = 1.0f / s;
  }

#pragma unroll
  for (int t = 0; t < 8; ++t) {
#pragma unroll
    for (int r = 0; r < 8; ++r) {
      Ps[(wave * 16 + r + hi * 8) * 128 + t * 16 + lhalf] = f2bf(acc[t][r] * rinv[r]);
    }
  }

  // ---- Phase 2: context = P * V, N=2048 in 128-wide chunks ----
  const int jg = (tid & 15) * 8;   // 8 consecutive j columns owned by thread
  const int er = (tid >> 4) * 2;   // even e-row base within a 32-row pass
#pragma unroll 1
  for (int jc = 0; jc < 16; ++jc) {
    const int j0 = jc * 128;
    __syncthreads();  // (also publishes Ps on first iteration)
    // stage Vt[j][e] <- v[e][j0+j]: vectorized 16B row loads, b32 scatter
#pragma unroll
    for (int p = 0; p < 4; ++p) {
      int e0 = er + p * 32;
      union { uint4 qv; unsigned short s[8]; } ra, rb;
      ra.qv = *(const uint4*)(v + base + (size_t)e0 * HWn + j0 + jg);
      rb.qv = *(const uint4*)(v + base + (size_t)(e0 + 1) * HWn + j0 + jg);
#pragma unroll
      for (int u = 0; u < 8; ++u) {
        unsigned int pk = (unsigned int)ra.s[u] | ((unsigned int)rb.s[u] << 16);
        *(unsigned int*)(Vt + (jg + u) * 128 + e0) = pk;
      }
    }
    __syncthreads();
#pragma unroll
    for (int nt = 0; nt < 8; ++nt) {
      v8f oc = zero8();
#pragma unroll
      for (int es = 0; es < 4; ++es) {
        v16bf af = ld_frag_a(Ps + (wave * 16 + lhalf) * 128 + es * 32 + hi * 8);
        v16bf bfr = ld_frag_b(Vt + (nt * 16 + lhalf) * 128 + es * 32 + hi * 8);
        oc = wmma_bf16(af, bfr, oc);
      }
#pragma unroll
      for (int r = 0; r < 8; ++r) {
        int row = wave * 16 + r + hi * 8;
        ctx[(size_t)(b * Cn + row) * HWn + h * Dn + j0 + nt * 16 + lhalf] = f2bf(oc[r]);
      }
    }
  }
}

// =====================================================================
// Kernel 3: 3x3 conv C->C as implicit GEMM. One block per (b, image
// row). M=co (8 tiles across 8 waves), N=128 pixels, K=1152 in 36
// chunks. im2col slice staged transposed ([x][k]) in LDS.
// =====================================================================
__global__ void __launch_bounds__(256) conv3x3_kernel(
    const unsigned short* __restrict__ ctx, const unsigned short* __restrict__ wbf,
    const float* __restrict__ bo, float* __restrict__ out) {
  const int tid = threadIdx.x;
  const int wave = tid >> 5;
  const int lane = tid & 31;
  const int lhalf = lane & 15;
  const int hi = lane >> 4;
  const int bx = blockIdx.x;
  const int b = bx >> 7;
  const int y = bx & 127;

  __shared__ __align__(32) unsigned short Bs[128 * 32];  // [x][k] bf16, 8KB

  v8f acc[8];
#pragma unroll
  for (int n = 0; n < 8; ++n) acc[n] = zero8();

#pragma unroll 1
  for (int ks = 0; ks < KW / 32; ++ks) {
    const int k0 = ks * 32;
    __syncthreads();
    {
      int kr = tid >> 3, seg = tid & 7;
      int kg = k0 + kr;
      int ci = kg / 9;
      int t = kg - ci * 9;
      int dy = t / 3 - 1;
      int dx = t - (t / 3) * 3 - 1;
      int yy = y + dy;
      const unsigned short* src = ctx + (size_t)(b * Cn + ci) * HWn + yy * Wn;
      bool yok = (yy >= 0) && (yy < Hn);
      for (int u = 0; u < 16; ++u) {
        int xx = seg * 16 + u;
        int xsrc = xx + dx;
        unsigned short val = 0;
        if (yok && xsrc >= 0 && xsrc < Wn) val = src[xsrc];
        Bs[xx * 32 + kr] = val;
      }
    }
    __syncthreads();
    // A-frag straight from global weights (K contiguous in wo layout)
    v16bf af = ld_frag_a(wbf + (size_t)(wave * 16 + lhalf) * KW + k0 + hi * 8);
#pragma unroll
    for (int nt = 0; nt < 8; ++nt) {
      v16bf bfr = ld_frag_b(Bs + (nt * 16 + lhalf) * 32 + hi * 8);
      acc[nt] = wmma_bf16(af, bfr, acc[nt]);
    }
  }

#pragma unroll
  for (int nt = 0; nt < 8; ++nt) {
#pragma unroll
    for (int r = 0; r < 8; ++r) {
      int co = wave * 16 + r + hi * 8;
      out[(size_t)(b * Cn + co) * HWn + y * Wn + nt * 16 + lhalf] = acc[nt][r] + bo[co];
    }
  }
}

// =====================================================================
// launch
// =====================================================================
extern "C" void kernel_launch(void* const* d_in, const int* in_sizes, int n_in,
                              void* d_out, int out_size, void* d_ws, size_t ws_size,
                              hipStream_t stream) {
  (void)in_sizes; (void)n_in; (void)out_size; (void)ws_size;
  const float* x  = (const float*)d_in[0];
  const float* wq = (const float*)d_in[1];
  const float* bq = (const float*)d_in[2];
  const float* wk = (const float*)d_in[3];
  const float* bk = (const float*)d_in[4];
  const float* wv = (const float*)d_in[5];
  const float* bv = (const float*)d_in[6];
  const float* wo = (const float*)d_in[7];
  const float* bo = (const float*)d_in[8];
  float* out = (float*)d_out;

  // workspace layout (bf16 intermediates): q | k | v | ctx | wo_bf16
  const size_t elems = (size_t)Bn * Cn * HWn;     // 16,777,216
  uint8_t* ws = (uint8_t*)d_ws;
  unsigned short* qb  = (unsigned short*)(ws);
  unsigned short* kb  = (unsigned short*)(ws + elems * 2);
  unsigned short* vb  = (unsigned short*)(ws + elems * 4);
  unsigned short* cxb = (unsigned short*)(ws + elems * 6);
  unsigned short* wob = (unsigned short*)(ws + elems * 8);
  const int wn = Cn * Cn * 9;                     // 147,456

  cvt_w_kernel<<<(wn + 255) / 256, 256, 0, stream>>>(wo, wob, wn);
  dwconv_qkv_kernel<<<dim3(16, Cn, Bn), 256, 0, stream>>>(
      x, wq, bq, wk, bk, wv, bv, qb, kb, vb);
  attn_kernel<<<Bn * HEADS, 256, 0, stream>>>(qb, kb, vb, cxb);
  conv3x3_kernel<<<Bn * Hn, 256, 0, stream>>>(cxb, wob, bo, out);
}
